// EncoderLocal_48206712931041
// MI455X (gfx1250) — matrix operand order repaired and checked by reading, not compiled, not run on hardware
//
#include <hip/hip_runtime.h>
#include <hip/hip_bf16.h>
#include <math.h>

// ---------------------------------------------------------------------------
// Problem constants (reference: B=64, S=4096, D=256, H=512, H2=256, ZL=16, W=64)
// ---------------------------------------------------------------------------
#define NSEQ   4096      // B*T independent LSTM sequences
#define DIN    256       // D
#define HID    512       // H
#define GATES  2048      // 4*H
#define H2DIM  256
#define ZLAT   16
#define TT     64        // T = S / window (also window length W numerically)

typedef __attribute__((ext_vector_type(16))) __bf16 v16bf;
typedef __attribute__((ext_vector_type(8)))  float  v8f;

union FragAB { v16bf v; unsigned int u[8]; };
union FragC  { v8f   v; float        f[8]; };

__device__ __forceinline__ unsigned short bf16u(float f) {
    union { __bf16 h; unsigned short u; } c; c.h = (__bf16)f;   // RNE fptrunc
    return c.u;
}
__device__ __forceinline__ unsigned int pkbf(float a, float b) {
    union { __bf16 h[2]; unsigned int u; } c;                   // -> v_cvt_pk_bf16_f32
    c.h[0] = (__bf16)a; c.h[1] = (__bf16)b;
    return c.u;
}
__device__ __forceinline__ float sigmoidf_(float x) {
    return 1.0f / (1.0f + __expf(-x));
}

// LDS byte-address of a __shared__ pointer (low 32 bits of the shared-aperture
// flat address are the wave-relative LDS offset).
__device__ __forceinline__ unsigned lds_addr(const void* p) {
    return (unsigned)(unsigned long long)(uintptr_t)p;
}

// ---- CDNA5 async global->LDS copy (ASYNCcnt), GVS mode: vdst_lds, voff, saddr
__device__ __forceinline__ void async_b128(unsigned lds_off, const void* base, unsigned byte_off) {
    asm volatile("global_load_async_to_lds_b128 %0, %1, %2"
                 :: "v"(lds_off), "v"(byte_off),
                    "s"((unsigned long long)(uintptr_t)base)
                 : "memory");
}
__device__ __forceinline__ void wait_async0() {
    asm volatile("s_wait_asynccnt 0" ::: "memory");
}

// A fragment (16x32 bf16): lane holds row (lane&15); K interleave
// {0..7,16..23} / {8..15,24..31} by lane half (ISA 7.12.2).
__device__ __forceinline__ v16bf frag_a_lds(const unsigned short* As, int rowBase, int lane) {
    const int row = rowBase + (lane & 15);
    const int hi  = (lane >> 4) * 8;
    FragAB fr;
#pragma unroll
    for (int p = 0; p < 8; ++p) {
        const int e = 2 * p;
        const int k = (e & 7) + ((e >> 3) << 4) + hi;
        fr.u[p] = *(const unsigned int*)(As + row * 32 + k);
    }
    return fr.v;
}
// B fragment (32x16 bf16): lane holds column; BsT staged col-major (stride 32 K).
__device__ __forceinline__ v16bf frag_b_lds(const unsigned short* BsT, int colBase, int lane) {
    const int n  = colBase + (lane & 15);
    const int hi = (lane >> 4) * 8;
    FragAB fr;
#pragma unroll
    for (int p = 0; p < 8; ++p) {
        const int k = ((2 * p) & 7) + (((2 * p) >> 3) << 4) + hi;
        fr.u[p] = *(const unsigned int*)(BsT + n * 32 + k);
    }
    return fr.v;
}

// ---------------------------------------------------------------------------
// One-time weight swizzlers: bf16, fragment-ready LDS order.
// lstm: [colBlock(8)][chunk(24)][gate(4)][col(64)][k(32)]   (3 MB)
// enc : [colBlock(4)][chunk(16)][col(64)][k(32)]            (256 KB)
// ---------------------------------------------------------------------------
__global__ __launch_bounds__(256) void swz_lstm_w(
    const float* __restrict__ Wx, const float* __restrict__ Wh,
    unsigned short* __restrict__ dst)
{
    const int idx = blockIdx.x * 256 + threadIdx.x;   // 1,572,864 total
    const int k  = idx & 31;
    const int c  = (idx >> 5) & 63;
    const int g  = (idx >> 11) & 3;
    const int r  = idx >> 13;               // cb*24 + ch
    const int ch = r % 24, cb = r / 24;
    const int kk = ch * 32 + k;
    const int col = g * HID + cb * 64 + c;
    const float v = (kk < DIN) ? Wx[(size_t)kk * GATES + col]
                               : Wh[(size_t)(kk - DIN) * GATES + col];
    dst[idx] = bf16u(v);
}

__global__ __launch_bounds__(256) void swz_enc_w(
    const float* __restrict__ W_enc, unsigned short* __restrict__ dst)
{
    const int idx = blockIdx.x * 256 + threadIdx.x;   // 131,072 total
    const int k  = idx & 31;
    const int c  = (idx >> 5) & 63;
    const int r  = idx >> 11;               // cb*16 + ch
    const int ch = r & 15, cb = r >> 4;
    dst[idx] = bf16u(W_enc[(size_t)(ch * 32 + k) * H2DIM + cb * 64 + c]);
}

// ---------------------------------------------------------------------------
// Kernel 1: one LSTM step.  Tile 128 rows x 64 hidden units (4 gate tiles).
// Double-buffered LDS + software-pipelined K-loop: chunk ch+1's async
// global->LDS copies are issued before the WMMA block for chunk ch, so the
// DMA overlaps the matrix math.  h kept bf16 across steps; c stays fp32.
// ---------------------------------------------------------------------------
__global__ __launch_bounds__(256) void lstm_step_kernel(
    const float* __restrict__ x,
    const unsigned short* __restrict__ h_in,   // bf16
    unsigned short* __restrict__ h_out,        // bf16
    float* __restrict__ c_buf,
    const unsigned short* __restrict__ wswz,   // pre-swizzled bf16 weights
    const float* __restrict__ b_lstm, int t)
{
    __shared__ __align__(16) unsigned short As[2 * 128 * 32];      // 2 x 8 KB
    __shared__ __align__(16) unsigned short BsT[2 * 4 * 64 * 32];  // 2 x 16 KB

    const int tid  = threadIdx.x;
    const int lane = tid & 31, wave = tid >> 5;
    const int rowBase = blockIdx.x * 128;
    const int colBase = blockIdx.y * 64;
    const bool first = (t == 0);

    const unsigned ldsA = lds_addr(As);
    const unsigned ldsB = lds_addr(BsT);

    FragC acc[4][4];
#pragma unroll
    for (int g = 0; g < 4; ++g)
#pragma unroll
        for (int f = 0; f < 4; ++f)
#pragma unroll
            for (int r = 0; r < 8; ++r) acc[g][f].f[r] = 0.0f;

    const int nChunks = first ? 8 : 24;      // t==0: h==0, skip recurrent half

    // stage chunk `c2` into buffer c2&1 (B always async; A async for h part,
    // fp32->bf16 packed convert for the x part)
    auto issue_chunk = [&](int c2) {
        const int kBase = c2 * 32;
        const int buf   = c2 & 1;
        {
            const unsigned short* wtile = wswz + ((size_t)blockIdx.y * 24 + c2) * 8192;
            const unsigned ldsBb = ldsB + (unsigned)buf * 16384u;
#pragma unroll
            for (int r = 0; r < 4; ++r) {
                const unsigned off = (unsigned)(tid + r * 256) * 16u;
                async_b128(ldsBb + off, wtile, off);
            }
        }
        if (kBase < DIN) {
            const int tq = tid & 7, tr = tid >> 3;
            unsigned short* Ab = As + buf * (128 * 32);
#pragma unroll
            for (int rr = 0; rr < 4; ++rr) {
                const int row  = tr + rr * 32;
                const int grow = rowBase + row;
                const float4 v = *(const float4*)(x + ((size_t)grow * TT + t) * DIN
                                                  + kBase + tq * 4);
                unsigned int* dst = (unsigned int*)(Ab + row * 32 + tq * 4);
                dst[0] = pkbf(v.x, v.y);
                dst[1] = pkbf(v.z, v.w);
            }
        } else {
            const unsigned ldsAb = ldsA + (unsigned)buf * 8192u;
#pragma unroll
            for (int p = 0; p < 2; ++p) {
                const int task = tid + p * 256;      // 512 tasks of 16 B
                const int row  = task >> 2;
                const int grp  = task & 3;
                const unsigned goff =
                    (unsigned)((rowBase + row) * HID + (kBase - DIN)) * 2u + grp * 16u;
                async_b128(ldsAb + (unsigned)row * 64u + grp * 16u, h_in, goff);
            }
        }
    };

    issue_chunk(0);
    for (int ch = 0; ch < nChunks; ++ch) {
        wait_async0();        // only chunk ch's copies are outstanding here
        __syncthreads();      // all waves have ch's data; all past compute(ch-1)
        if (ch + 1 < nChunks) issue_chunk(ch + 1);   // overlaps with WMMAs below

        const int buf = ch & 1;
        const unsigned short* Ab = As  + buf * (128 * 32);
        const unsigned short* Bb = BsT + buf * (4 * 64 * 32);
        const v16bf a = frag_a_lds(Ab, wave * 16, lane);
#pragma unroll
        for (int g = 0; g < 4; ++g) {
#pragma unroll
            for (int f = 0; f < 4; ++f) {
                const v16bf b = frag_b_lds(Bb + g * 64 * 32, f * 16, lane);
                acc[g][f].v = __builtin_amdgcn_wmma_f32_16x16x32_bf16(
                    false, a, false, b, (short)0, acc[g][f].v, false, false);
            }
        }
    }

    // ---- gate nonlinearity + state update (reference uses all-sigmoid)
    const int hi = (lane >> 4) * 8, n = lane & 15;
#pragma unroll
    for (int f = 0; f < 4; ++f) {
        const int col = colBase + f * 16 + n;
        const float bi  = b_lstm[col];
        const float bff = b_lstm[HID + col];
        const float bg  = b_lstm[2 * HID + col];
        const float bo  = b_lstm[3 * HID + col];
#pragma unroll
        for (int r = 0; r < 8; ++r) {
            const int grow = rowBase + wave * 16 + r + hi;
            const size_t idx = (size_t)grow * HID + col;
            const float ig = sigmoidf_(acc[0][f].f[r] + bi);
            const float fg = sigmoidf_(acc[1][f].f[r] + bff);
            const float gg = sigmoidf_(acc[2][f].f[r] + bg);
            const float og = sigmoidf_(acc[3][f].f[r] + bo);
            const float cold = first ? 0.0f : c_buf[idx];
            const float cnew = fg * cold + ig * gg;
            c_buf[idx] = cnew;
            h_out[idx] = bf16u(og * sigmoidf_(cnew));
        }
    }
}

// ---------------------------------------------------------------------------
// Kernel 2: feat = relu(h @ W_enc + b_enc).  Fully async-staged, same
// double-buffered pipeline as the step kernel.
// ---------------------------------------------------------------------------
__global__ __launch_bounds__(256) void feat_kernel(
    const unsigned short* __restrict__ h,       // bf16
    const unsigned short* __restrict__ wenc,    // pre-swizzled bf16
    const float* __restrict__ b_enc, float* __restrict__ feat)
{
    __shared__ __align__(16) unsigned short As[2 * 128 * 32];
    __shared__ __align__(16) unsigned short BsT[2 * 64 * 32];

    const int tid = threadIdx.x, lane = tid & 31, wave = tid >> 5;
    const int rowBase = blockIdx.x * 128;
    const int colBase = blockIdx.y * 64;
    const unsigned ldsA = lds_addr(As);
    const unsigned ldsB = lds_addr(BsT);

    FragC acc[4];
#pragma unroll
    for (int f = 0; f < 4; ++f)
#pragma unroll
        for (int r = 0; r < 8; ++r) acc[f].f[r] = 0.0f;

    auto issue_chunk = [&](int c2) {
        const int kBase = c2 * 32;
        const int buf   = c2 & 1;
        {   // B tile: 4 KB contiguous
            const unsigned short* wtile = wenc + ((size_t)blockIdx.y * 16 + c2) * 2048;
            const unsigned off = (unsigned)tid * 16u;
            async_b128(ldsB + (unsigned)buf * 4096u + off, wtile, off);
        }
        {   // A tile: bf16 h rows
            const unsigned ldsAb = ldsA + (unsigned)buf * 8192u;
#pragma unroll
            for (int p = 0; p < 2; ++p) {
                const int task = tid + p * 256;
                const int row  = task >> 2;
                const int grp  = task & 3;
                const unsigned goff =
                    (unsigned)((rowBase + row) * HID + kBase) * 2u + grp * 16u;
                async_b128(ldsAb + (unsigned)row * 64u + grp * 16u, h, goff);
            }
        }
    };

    issue_chunk(0);
    for (int ch = 0; ch < HID / 32; ++ch) {
        wait_async0();
        __syncthreads();
        if (ch + 1 < HID / 32) issue_chunk(ch + 1);

        const int buf = ch & 1;
        const unsigned short* Ab = As  + buf * (128 * 32);
        const unsigned short* Bb = BsT + buf * (64 * 32);
        const v16bf a = frag_a_lds(Ab, wave * 16, lane);
#pragma unroll
        for (int f = 0; f < 4; ++f) {
            const v16bf b = frag_b_lds(Bb, f * 16, lane);
            acc[f].v = __builtin_amdgcn_wmma_f32_16x16x32_bf16(
                false, a, false, b, (short)0, acc[f].v, false, false);
        }
    }

    const int hi = (lane >> 4) * 8, n = lane & 15;
#pragma unroll
    for (int f = 0; f < 4; ++f) {
        const int col = colBase + f * 16 + n;
        const float bb = b_enc[col];
#pragma unroll
        for (int r = 0; r < 8; ++r) {
            const int grow = rowBase + wave * 16 + r + hi;
            const float v = acc[f].f[r] + bb;
            feat[(size_t)grow * H2DIM + col] = v > 0.0f ? v : 0.0f;
        }
    }
}

// ---------------------------------------------------------------------------
// Kernel 3: heads. mean -> d_out in (B,ZL,T) layout; softplus(std) -> ws.
// 48 output cols = 3 fragments (mean 0:16, cov 0:16, cov 16:32).
// ---------------------------------------------------------------------------
__global__ __launch_bounds__(256) void heads_kernel(
    const float* __restrict__ feat,
    const float* __restrict__ W_mean, const float* __restrict__ b_mean,
    const float* __restrict__ W_cov,  const float* __restrict__ b_cov,
    float* __restrict__ out_mean, float* __restrict__ stdbuf)
{
    __shared__ __align__(16) unsigned short As[128 * 32];
    __shared__ __align__(16) unsigned short BsT[48 * 32];

    const int tid = threadIdx.x, lane = tid & 31, wave = tid >> 5;
    const int rowBase = blockIdx.x * 128;

    FragC acc[3];
#pragma unroll
    for (int f = 0; f < 3; ++f)
#pragma unroll
        for (int r = 0; r < 8; ++r) acc[f].f[r] = 0.0f;

    for (int ch = 0; ch < H2DIM / 32; ++ch) {
        const int kBase = ch * 32;
        __syncthreads();
        {   // A: fp32 feat -> packed bf16
            const int tq = tid & 7, tr = tid >> 3;
#pragma unroll
            for (int rr = 0; rr < 4; ++rr) {
                const int row = tr + rr * 32;
                const int grow = rowBase + row;
                const float4 v = *(const float4*)(feat + (size_t)grow * H2DIM + kBase + tq * 4);
                unsigned int* dst = (unsigned int*)(As + row * 32 + tq * 4);
                dst[0] = pkbf(v.x, v.y);
                dst[1] = pkbf(v.z, v.w);
            }
        }
        {   // B: 48 cols scatter (small)
            const int kk = tid & 31;
            const int cg = (tid >> 5) * 6;
#pragma unroll
            for (int cc = 0; cc < 6; ++cc) {
                const int c = cg + cc;
                const float v = (c < ZLAT)
                    ? W_mean[(size_t)(kBase + kk) * ZLAT + c]
                    : W_cov[(size_t)(kBase + kk) * (2 * ZLAT) + (c - ZLAT)];
                BsT[c * 32 + kk] = bf16u(v);
            }
        }
        __syncthreads();
        const v16bf a = frag_a_lds(As, wave * 16, lane);
#pragma unroll
        for (int f = 0; f < 3; ++f) {
            const v16bf b = frag_b_lds(BsT, f * 16, lane);
            acc[f].v = __builtin_amdgcn_wmma_f32_16x16x32_bf16(
                false, a, false, b, (short)0, acc[f].v, false, false);
        }
    }

    const int hi = (lane >> 4) * 8, n = lane & 15;
    {   // mean -> (B, ZL, T)
        const float bb = b_mean[n];
#pragma unroll
        for (int r = 0; r < 8; ++r) {
            const int grow = rowBase + wave * 16 + r + hi;   // = b*T + tt
            const int bidx = grow >> 6, tt = grow & 63;
            out_mean[(size_t)bidx * (ZLAT * TT) + n * TT + tt] = acc[0].f[r] + bb;
        }
    }
#pragma unroll
    for (int q = 0; q < 2; ++q) {   // softplus(std) -> ws (row-major 4096x32)
        const int c = q * ZLAT + n;
        const float bb = b_cov[c];
#pragma unroll
        for (int r = 0; r < 8; ++r) {
            const int grow = rowBase + wave * 16 + r + hi;
            const float xv = acc[1 + q].f[r] + bb;
            const float sp = (xv > 20.0f) ? xv : log1pf(__expf(xv));
            stdbuf[(size_t)grow * (2 * ZLAT) + c] = sp;
        }
    }
}

// ---------------------------------------------------------------------------
// Kernel 4: closed-form inverse of the upper-bidiagonal precision matrix,
// written transposed as lower-triangular scale_tril, isfinite->0 masked.
// ---------------------------------------------------------------------------
__global__ __launch_bounds__(64) void tril_kernel(
    const float* __restrict__ stdbuf, float* __restrict__ out_scale)
{
    const int p = blockIdx.x;            // b*ZL + z
    const int b = p >> 4, z = p & 15;
    const int j = threadIdx.x;

    __shared__ float d[TT], s[TT];
    {
        const float* row = stdbuf + (size_t)(b * TT + j) * (2 * ZLAT);
        d[j] = row[2 * z] + 1.0f;
        s[j] = row[2 * z + 1];
    }
    __syncthreads();

    float* dst = out_scale + ((size_t)p * TT + j) * TT;
    for (int i = j + 1; i < TT; ++i) dst[i] = 0.0f;

    float prev = 0.0f;
    for (int i = j; i >= 0; --i) {
        const float v = (i == j) ? (1.0f / d[i]) : (-s[i] * prev / d[i]);
        dst[i] = __builtin_isfinite(v) ? v : 0.0f;
        prev = v;
    }
}

// ---------------------------------------------------------------------------
extern "C" void kernel_launch(void* const* d_in, const int* in_sizes, int n_in,
                              void* d_out, int out_size, void* d_ws, size_t ws_size,
                              hipStream_t stream) {
    const float* x      = (const float*)d_in[0];
    // d_in[1] = window_size (hardcoded 64)
    const float* Wx     = (const float*)d_in[2];
    const float* Wh     = (const float*)d_in[3];
    const float* b_lstm = (const float*)d_in[4];
    const float* W_enc  = (const float*)d_in[5];
    const float* b_enc  = (const float*)d_in[6];
    const float* W_mean = (const float*)d_in[7];
    const float* b_mean = (const float*)d_in[8];
    const float* W_cov  = (const float*)d_in[9];
    const float* b_cov  = (const float*)d_in[10];

    // ---- workspace carve-up (~24 MB, all 16-B aligned) ----
    float* ws   = (float*)d_ws;
    float* cB   = ws;                                   // 4096*512 f32
    float* feat = cB + (size_t)NSEQ * HID;              // 4096*256 f32
    float* stdb = feat + (size_t)NSEQ * H2DIM;          // 4096*32  f32
    unsigned short* hA   = (unsigned short*)(stdb + (size_t)NSEQ * 2 * ZLAT);
    unsigned short* hB   = hA + (size_t)NSEQ * HID;     // bf16 h ping/pong
    unsigned short* wswz = hB + (size_t)NSEQ * HID;     // 1,572,864 bf16
    unsigned short* wenc = wswz + (size_t)8 * 24 * 8192; // 131,072 bf16

    float* out_mean  = (float*)d_out;                   // 64*16*64
    float* out_scale = out_mean + (size_t)64 * ZLAT * TT;

    const dim3 blk(256);

    // one-time weight swizzles (bf16, fragment-ready)
    hipLaunchKernelGGL(swz_lstm_w, dim3(6144), blk, 0, stream, Wx, Wh, wswz);
    hipLaunchKernelGGL(swz_enc_w,  dim3(512),  blk, 0, stream, W_enc, wenc);

    const dim3 stepGrid(NSEQ / 128, (GATES / 4) / 64);  // (32, 8)
    for (int t = 0; t < TT; ++t) {
        const unsigned short* hin = (t & 1) ? hB : hA;  // t==0 reads nothing
        unsigned short* hout      = (t & 1) ? hA : hB;  // t==63 writes hA
        hipLaunchKernelGGL(lstm_step_kernel, stepGrid, blk, 0, stream,
                           x, hin, hout, cB, wswz, b_lstm, t);
    }
    hipLaunchKernelGGL(feat_kernel, dim3(NSEQ / 128, H2DIM / 64), blk, 0, stream,
                       hA, wenc, b_enc, feat);
    hipLaunchKernelGGL(heads_kernel, dim3(NSEQ / 128), blk, 0, stream,
                       feat, W_mean, b_mean, W_cov, b_cov, out_mean, stdb);
    hipLaunchKernelGGL(tril_kernel, dim3(64 * ZLAT), dim3(TT), 0, stream,
                       stdb, out_scale);
}
